// inputEncoder_10565619548659
// MI455X (gfx1250) — matrix operand order repaired and checked by reading, not compiled, not run on hardware
//
#include <hip/hip_runtime.h>

// MI455X / gfx1250, wave32. Fused  sim = V·Vᵀ ; norm_sim = sim/(||row||+eps) ;
// out = norm_sim · T   using v_wmma_f32_16x16x32_f16 (f16 in, f32 accumulate).
//
// Workspace layout:
//   [0, 2MB)    Vh : f16 copy of encoder_vector  [8][2048][64]
//   [2MB,18MB)  Tt : f16 transposed encoder_T    [8][512][2048]  (Tt[b][n][j])

#define EPS_F 1e-10f

typedef _Float16 v16h __attribute__((ext_vector_type(16)));
typedef _Float16 v8h  __attribute__((ext_vector_type(8)));
typedef float    v8f  __attribute__((ext_vector_type(8)));

// ---------------- prep: f32 -> f16 copy of V ----------------
__global__ void conv_v_kernel(const float* __restrict__ V,
                              _Float16* __restrict__ Vh, int n) {
  int i = blockIdx.x * blockDim.x + threadIdx.x;
  if (i < n) Vh[i] = (_Float16)V[i];
}

// ---------------- prep: f32 -> f16 transpose of T ----------------
// T  flat: [8][2048][512]   (b, j, n)
// Tt flat: [8][512][2048]   (b, n, j)  -> B-operand reads are contiguous in j
__global__ void conv_t_kernel(const float* __restrict__ T,
                              _Float16* __restrict__ Tt) {
  long idx = (long)blockIdx.x * blockDim.x + threadIdx.x;
  const long total = 8L * 512 * 2048;
  if (idx >= total) return;
  long j = idx & 2047;
  long n = (idx >> 11) & 511;
  long b = idx >> 20;
  Tt[idx] = (_Float16)T[(b * 2048 + j) * 512 + n];
}

// ---------------- fused main kernel ----------------
// grid = 8 batches * 128 i-tiles; block = 256 threads = 8 waves.
// Each workgroup: 16 output rows (i0..i0+15), all 512 output columns.
__global__ __launch_bounds__(256)
void fused_norm_gemm_kernel(const _Float16* __restrict__ Vh,
                            const _Float16* __restrict__ Tt,
                            float* __restrict__ out) {
  __shared__ float s_sumsq[16];
  __shared__ float s_scale[16];
  // scaled f16 sim tiles, A-operand friendly: Sh[jt][row i][col j-in-chunk]
  __shared__ alignas(32) _Float16 Sh[8][16][32];

  const int b   = blockIdx.x >> 7;           // /128
  const int i0  = (blockIdx.x & 127) << 4;   // 16-row tile
  const int tid = threadIdx.x;
  const int w   = tid >> 5;                  // wave id 0..7
  const int l   = tid & 31;                  // lane
  const int lm  = l & 15;
  const int hhi = l >> 4;                    // 0 = lanes 0-15, 1 = lanes 16-31

  if (tid < 16) s_sumsq[tid] = 0.0f;
  __syncthreads();

  const _Float16* Vb  = Vh + (long)b * 2048 * 64;
  const _Float16* Ttb = Tt + (long)b * 512 * 2048;

  // --- preload A operand: V rows i0..i0+15, K=64 as two 16x32 f16 slices.
  // A k order used here (contiguous, same as B): lane holds row M=lane%16; halves 0-15 = K 2*base8..2*base8+15,
  // base8 = (lane<16 ? 0 : 8).
  v16h a0, a1;
  {
    const _Float16* arow = Vb + (long)(i0 + lm) * 64;
    const int base8 = hhi ? 8 : 0;
    union { v16h v; v8h h[2]; } u0, u1;
    u0.h[0] = *(const v8h*)(arow + 2 * base8);
    u0.h[1] = *(const v8h*)(arow + 8 + 2 * base8);
    u1.h[0] = *(const v8h*)(arow + 32 + 2 * base8);
    u1.h[1] = *(const v8h*)(arow + 40 + 2 * base8);
    a0 = u0.v;
    a1 = u1.v;
  }
  // B k order used here (contiguous, same as A): lane holds column N=lane%16, 16 contiguous K halves
  // starting at (lane<16 ? 0 : 16).
  const int bsel = hhi ? 16 : 0;

  // ================= pass 1: row sums of sim^2 =================
  // wave w covers j in [w*256, (w+1)*256)
  float ss[8];
#pragma unroll
  for (int r = 0; r < 8; ++r) ss[r] = 0.0f;

  for (int t = 0; t < 8; ++t) {
    const int j0 = (w << 8) + (t << 5);
#pragma unroll
    for (int half = 0; half < 2; ++half) {
      const _Float16* brow = Vb + (long)(j0 + half * 16 + lm) * 64;
      v16h b0 = *(const v16h*)(brow + bsel);        // K = 0..31
      v16h b1 = *(const v16h*)(brow + 32 + bsel);   // K = 32..63
      v8f d = {};
      d = __builtin_amdgcn_wmma_f32_16x16x32_f16(false, a0, false, b0,
                                                 (short)0, d, false, false);
      d = __builtin_amdgcn_wmma_f32_16x16x32_f16(false, a1, false, b1,
                                                 (short)0, d, false, false);
      // D layout: VGPR r holds row M = r + 8*hhi, lane column N = lane%16
#pragma unroll
      for (int r = 0; r < 8; ++r) ss[r] += d[r] * d[r];
    }
  }
  // reduce over the 16 lanes of each half (columns of the tile)
#pragma unroll
  for (int m = 8; m >= 1; m >>= 1) {
#pragma unroll
    for (int r = 0; r < 8; ++r) ss[r] += __shfl_xor(ss[r], m, 32);
  }
  if (lm == 0) {
#pragma unroll
    for (int r = 0; r < 8; ++r) atomicAdd(&s_sumsq[r + 8 * hhi], ss[r]);
  }
  __syncthreads();
  if (tid < 16) s_scale[tid] = 1.0f / (sqrtf(s_sumsq[tid]) + EPS_F);
  __syncthreads();

  float sc[8];
#pragma unroll
  for (int r = 0; r < 8; ++r) sc[r] = s_scale[r + 8 * hhi];

  // ================= pass 2: out = norm_sim @ T =================
  v8f acc[4] = {};               // 4 n-tiles of 16 cols -> 64 cols per wave
  const int n0w = w << 6;        // wave's n base

  for (int c = 0; c < 8; ++c) {  // j chunks of 256
    const int jc = c << 8;
    // this wave recomputes + scales its 16x32 sim tile into LDS
    {
      const int j0 = jc + (w << 5);
#pragma unroll
      for (int half = 0; half < 2; ++half) {
        const _Float16* brow = Vb + (long)(j0 + half * 16 + lm) * 64;
        v16h b0 = *(const v16h*)(brow + bsel);
        v16h b1 = *(const v16h*)(brow + 32 + bsel);
        v8f d = {};
        d = __builtin_amdgcn_wmma_f32_16x16x32_f16(false, a0, false, b0,
                                                   (short)0, d, false, false);
        d = __builtin_amdgcn_wmma_f32_16x16x32_f16(false, a1, false, b1,
                                                   (short)0, d, false, false);
#pragma unroll
        for (int r = 0; r < 8; ++r) {
          Sh[w][r + 8 * hhi][half * 16 + lm] = (_Float16)(d[r] * sc[r]);
        }
      }
    }
    __syncthreads();

#pragma unroll
    for (int jt = 0; jt < 8; ++jt) {
      // A' (norm_sim 16x32 tile) from LDS in A-operand layout
      union { v16h v; v8h h[2]; } ua;
      const _Float16* srow = &Sh[jt][lm][0];
      const int base8 = hhi ? 8 : 0;
      ua.h[0] = *(const v8h*)(srow + 2 * base8);
      ua.h[1] = *(const v8h*)(srow + 8 + 2 * base8);
      const v16h ap = ua.v;
      const int jglob = jc + (jt << 5);
#pragma unroll
      for (int nt = 0; nt < 4; ++nt) {
        const _Float16* trow =
            Ttb + (long)(n0w + nt * 16 + lm) * 2048 + jglob + bsel;
        v16h bp = *(const v16h*)trow;   // one aligned 32B load per WMMA
        acc[nt] = __builtin_amdgcn_wmma_f32_16x16x32_f16(
            false, ap, false, bp, (short)0, acc[nt], false, false);
      }
    }
    __syncthreads();
  }

  // ---- write out: out[b][i0+M][n], M = r + 8*hhi, N = lane%16 + 16*nt
  float* outb = out + ((long)b * 2048 + i0) * 512;
#pragma unroll
  for (int nt = 0; nt < 4; ++nt) {
    const int ncol = n0w + nt * 16 + lm;
#pragma unroll
    for (int r = 0; r < 8; ++r) {
      outb[(long)(r + 8 * hhi) * 512 + ncol] = acc[nt][r];
    }
  }
}

// ---------------- launcher ----------------
extern "C" void kernel_launch(void* const* d_in, const int* in_sizes, int n_in,
                              void* d_out, int out_size, void* d_ws,
                              size_t ws_size, hipStream_t stream) {
  const float* V = (const float*)d_in[0];   // [8][2048][64]
  const float* T = (const float*)d_in[1];   // [8][2048][32][16]
  float* out = (float*)d_out;               // [8][2048][512]

  _Float16* Vh = (_Float16*)d_ws;                                   // 2 MB
  _Float16* Tt = (_Float16*)((char*)d_ws + (size_t)8 * 2048 * 64 * 2); // 16 MB

  const int nV = 8 * 2048 * 64;
  conv_v_kernel<<<(nV + 255) / 256, 256, 0, stream>>>(V, Vh, nV);

  const long nT = 8L * 512 * 2048;
  conv_t_kernel<<<(int)((nT + 255) / 256), 256, 0, stream>>>(T, Tt);

  fused_norm_gemm_kernel<<<8 * 128, 256, 0, stream>>>(Vh, Tt, out);
}
